// WeightNet_42949672960313
// MI455X (gfx1250) — matrix operand order repaired and checked by reading, not compiled, not run on hardware
//
#include <hip/hip_runtime.h>
#include <hip/hip_bf16.h>

#define NL 21      // number of kernel positions
#define C0 32
#define C1 32
#define IN_DIM 7
#define WAVES 8    // waves per block in the GEMM kernel (256 threads, wave32)

typedef __attribute__((ext_vector_type(16))) _Float16 v16h;
typedef __attribute__((ext_vector_type(8)))  _Float16 v8h;
typedef __attribute__((ext_vector_type(8)))  float    v8f;

// ---------------------------------------------------------------------------
// Pass 1: zero counters
// ---------------------------------------------------------------------------
__global__ void wn_zero(int* counts, int* cursors) {
    int t = threadIdx.x;
    if (t < NL) { counts[t] = 0; cursors[t] = 0; }
}

// ---------------------------------------------------------------------------
// Pass 2: LDS-staged histogram of idx
// ---------------------------------------------------------------------------
__global__ __launch_bounds__(256) void wn_hist(const int* __restrict__ idx, int E,
                                               int* __restrict__ counts) {
    __shared__ int lc[NL];
    if (threadIdx.x < NL) lc[threadIdx.x] = 0;
    __syncthreads();
    for (int i = blockIdx.x * blockDim.x + threadIdx.x; i < E;
         i += gridDim.x * blockDim.x) {
        atomicAdd(&lc[idx[i]], 1);
    }
    __syncthreads();
    if (threadIdx.x < NL && lc[threadIdx.x]) atomicAdd(&counts[threadIdx.x], lc[threadIdx.x]);
}

// ---------------------------------------------------------------------------
// Pass 3: tiny exclusive scan with each bin padded to a multiple of 16 slots
// ---------------------------------------------------------------------------
__global__ void wn_scan(const int* __restrict__ counts, int* __restrict__ offs) {
    if (threadIdx.x == 0) {
        int acc = 0;
        for (int b = 0; b < NL; ++b) {
            offs[b] = acc;
            acc += (counts[b] + 15) & ~15;   // 16-row tile alignment per bin
        }
    }
}

// ---------------------------------------------------------------------------
// Pass 4: sentinel-fill the padded sorted-id buffer
// ---------------------------------------------------------------------------
__global__ __launch_bounds__(256) void wn_init(int* __restrict__ sorted, int cap) {
    int i = blockIdx.x * blockDim.x + threadIdx.x;
    if (i < cap) sorted[i] = -1;
}

// ---------------------------------------------------------------------------
// Pass 5: LDS-staged counting-sort scatter of edge ids into bin regions
// ---------------------------------------------------------------------------
__global__ __launch_bounds__(256) void wn_scatter(const int* __restrict__ idx, int E,
                                                  const int* __restrict__ offs,
                                                  int* __restrict__ cursors,
                                                  int* __restrict__ sorted) {
    __shared__ int lc[NL];
    __shared__ int base[NL];
    int t = threadIdx.x;
    if (t < NL) lc[t] = 0;
    __syncthreads();
    int i = blockIdx.x * blockDim.x + t;
    int b = -1, rank = 0;
    if (i < E) { b = idx[i]; rank = atomicAdd(&lc[b], 1); }
    __syncthreads();
    if (t < NL) base[t] = lc[t] ? atomicAdd(&cursors[t], lc[t]) : 0;
    __syncthreads();
    if (i < E) sorted[offs[b] + base[b] + rank] = i;
}

// ---------------------------------------------------------------------------
// WMMA helpers (16-bit operand layouts per CDNA5 ISA 7.12.2, wave32)
// ---------------------------------------------------------------------------
__device__ __forceinline__ v16h wn_make16(v8h lo, v8h hi) {
    return __builtin_shufflevector(lo, hi, 0, 1, 2, 3, 4, 5, 6, 7,
                                           8, 9, 10, 11, 12, 13, 14, 15);
}

// B operand (32x16 f16, K x N): lane n = lane%16, K = 8*half.. and 16+8*half..
// Weights stored transposed in LDS as [c][k] (32 x 32 halves, 64B rows), so
// a lane's 16 K-values are two contiguous 16B chunks -> two ds_load_b128.
__device__ __forceinline__ v16h wn_loadB(const _Float16* wt, int n, int half) {
    const v8h* row = (const v8h*)(wt + n * 32);
    return wn_make16(row[half], row[half + 2]);
}

__device__ __forceinline__ float wn_lrelu(float x) {
    return x >= 0.0f ? x : 0.2f * x;
}

// ---------------------------------------------------------------------------
// Pass 6: tiled two-layer MLP, one wave32 per 16-edge idx-homogeneous tile
// ---------------------------------------------------------------------------
__global__ __launch_bounds__(256) void wn_gemm(const float* __restrict__ inp,
                                               const int*   __restrict__ g_idx,
                                               const int*   __restrict__ sorted,
                                               int nTiles,
                                               const float* __restrict__ W0,
                                               const float* __restrict__ b0,
                                               const float* __restrict__ W1,
                                               const float* __restrict__ b1,
                                               float* __restrict__ out) {
    extern __shared__ char smem[];
    _Float16* w0t    = (_Float16*)smem;                    // NL*32*32 f16 (K padded 7->32)
    _Float16* w1t    = w0t + NL * 1024;                    // NL*32*32 f16
    float*    sb0    = (float*)(w1t + NL * 1024);          // NL*32
    float*    sb1    = sb0 + NL * 32;                      // NL*32
    _Float16* hstage = (_Float16*)(sb1 + NL * 32);         // WAVES*16*32 f16
    int*      eids   = (int*)(hstage + WAVES * 16 * 32);   // WAVES*16

    const int tid = threadIdx.x;

    // Cooperative load of all 21 weight sets, transposed to [l][c][k] in f16.
    for (int i = tid; i < NL * 1024; i += 256) {
        int l = i >> 10, r = i & 1023, c = r >> 5, k = r & 31;
        w0t[i] = (k < IN_DIM) ? (_Float16)W0[(l * IN_DIM + k) * C0 + c] : (_Float16)0.0f;
        w1t[i] = (_Float16)W1[(l * C0 + k) * C1 + c];
    }
    for (int i = tid; i < NL * 32; i += 256) { sb0[i] = b0[i]; sb1[i] = b1[i]; }
    __syncthreads();

    const int wave = tid >> 5;
    const int lane = tid & 31;
    const int m16  = lane & 15;   // row (A) or column (B/D) index
    const int half = lane >> 4;   // lane-half selects K block

    const int tile = blockIdx.x * WAVES + wave;
    if (tile >= nTiles) return;

    int* myEids = eids + wave * 16;
    if (lane < 16) myEids[lane] = sorted[tile * 16 + lane];
    asm volatile("s_wait_dscnt 0" ::: "memory");

    // Bins are 16-aligned and filled from slot 0, so slot 0 valid <=> tile non-empty.
    const int e0 = myEids[0];
    if (e0 < 0) return;                 // fully padded tile
    const int t_idx = g_idx[e0];        // uniform across the tile by construction
    const int myEdge = myEids[m16];     // this lane's A-row edge (may be -1 pad)

    // ---- Layer 0: A = 16x32 f16 gather of inputs (K 7..31 zero-padded) ----
    v16h a0;
#pragma unroll
    for (int e = 0; e < 16; ++e) {
        const int k = e + half * 8 + ((e >= 8) ? 8 : 0);   // ISA A-matrix 16x32 layout
        float v = 0.0f;
        if (myEdge >= 0 && k < IN_DIM) v = inp[(size_t)myEdge * IN_DIM + k];
        a0[e] = (_Float16)v;
    }

    const _Float16* w0l = w0t + t_idx * 1024;
    v16h bm0 = wn_loadB(w0l, m16,      half);
    v16h bm1 = wn_loadB(w0l, m16 + 16, half);

    v8f acc0 = {}, acc1 = {};
    acc0 = __builtin_amdgcn_wmma_f32_16x16x32_f16(false, a0, false, bm0, (short)0, acc0, false, false);
    acc1 = __builtin_amdgcn_wmma_f32_16x16x32_f16(false, a0, false, bm1, (short)0, acc1, false, false);

    // ---- Bias + leaky-ReLU, stage h (16x32) in LDS for operand relayout ----
    _Float16* hst = hstage + wave * 16 * 32;
#pragma unroll
    for (int j = 0; j < 2; ++j) {
        const int c = m16 + 16 * j;
        const float bias = sb0[t_idx * 32 + c];
#pragma unroll
        for (int r = 0; r < 8; ++r) {
            const int m = r + half * 8;                    // ISA C/D 16x16 layout
            const float x = wn_lrelu((j ? acc1[r] : acc0[r]) + bias);
            hst[m * 32 + c] = (_Float16)x;
        }
    }
    asm volatile("s_wait_dscnt 0" ::: "memory");

    // ---- Layer 1: reload h as A operand (two contiguous v8h per lane) ----
    const v8h* hrow = (const v8h*)(hst + m16 * 32);
    v16h a1 = wn_make16(hrow[half], hrow[2 + half]);       // K = half*8.., 16+half*8..

    const _Float16* w1l = w1t + t_idx * 1024;
    v16h bn0 = wn_loadB(w1l, m16,      half);
    v16h bn1 = wn_loadB(w1l, m16 + 16, half);

    v8f d0 = {}, d1 = {};
    d0 = __builtin_amdgcn_wmma_f32_16x16x32_f16(false, a1, false, bn0, (short)0, d0, false, false);
    d1 = __builtin_amdgcn_wmma_f32_16x16x32_f16(false, a1, false, bn1, (short)0, d1, false, false);

    // ---- Bias + leaky-ReLU, scatter 32 f32 per valid edge ----
    const float bias0 = sb1[t_idx * 32 + m16];
    const float bias1 = sb1[t_idx * 32 + m16 + 16];
#pragma unroll
    for (int r = 0; r < 8; ++r) {
        const int m = r + half * 8;
        const int edge = myEids[m];
        if (edge < 0) continue;
        float x0 = wn_lrelu(d0[r] + bias0);
        float x1 = wn_lrelu(d1[r] + bias1);
        out[(size_t)edge * C1 + m16]      = x0;
        out[(size_t)edge * C1 + m16 + 16] = x1;
    }
}

// ---------------------------------------------------------------------------
// Host launcher
// ---------------------------------------------------------------------------
extern "C" void kernel_launch(void* const* d_in, const int* in_sizes, int n_in,
                              void* d_out, int out_size, void* d_ws, size_t ws_size,
                              hipStream_t stream) {
    const float* inp = (const float*)d_in[0];
    const int*   idx = (const int*)d_in[1];
    const float* W0  = (const float*)d_in[2];
    const float* b0  = (const float*)d_in[3];
    const float* W1  = (const float*)d_in[4];
    const float* b1  = (const float*)d_in[5];
    float*       out = (float*)d_out;

    const int E = in_sizes[1];                       // number of edges

    int* ws      = (int*)d_ws;
    int* counts  = ws;                               // 21 (padded to 32)
    int* cursors = ws + 32;                          // 21
    int* offs    = ws + 64;                          // 21
    int* sorted  = ws + 96;                          // capacity ints

    const int capacity = ((E + 15) & ~15) + NL * 16; // worst-case per-bin padding
    const int nTiles   = capacity / 16;

    wn_zero<<<1, 64, 0, stream>>>(counts, cursors);
    wn_hist<<<256, 256, 0, stream>>>(idx, E, counts);
    wn_scan<<<1, 32, 0, stream>>>(counts, offs);
    wn_init<<<(capacity + 255) / 256, 256, 0, stream>>>(sorted, capacity);
    wn_scatter<<<(E + 255) / 256, 256, 0, stream>>>(idx, E, offs, cursors, sorted);

    constexpr size_t smemBytes =
        (size_t)NL * 1024 * 2 * sizeof(_Float16)     // w0t + w1t
        + (size_t)NL * 32 * 2 * sizeof(float)        // sb0 + sb1
        + (size_t)WAVES * 16 * 32 * sizeof(_Float16) // hstage
        + (size_t)WAVES * 16 * sizeof(int);          // eids
    (void)hipFuncSetAttribute((const void*)wn_gemm,
                              hipFuncAttributeMaxDynamicSharedMemorySize,
                              (int)smemBytes);

    wn_gemm<<<(nTiles + WAVES - 1) / WAVES, 256, smemBytes, stream>>>(
        inp, idx, sorted, nTiles, W0, b0, W1, b1, out);
}
